// GPT_62766652063920
// MI455X (gfx1250) — matrix operand order repaired and checked by reading, not compiled, not run on hardware
//
#include <hip/hip_runtime.h>
#include <hip/hip_bf16.h>
#include <stdint.h>

#define TT 1024
#define CC 768
#define HH 12
#define DD 64
#define LLAYERS 12
#define VV 50257
#define BBATCH 2
#define MTOK (BBATCH * TT) /* 2048 rows */

typedef __attribute__((ext_vector_type(16))) __bf16 v16bf;
typedef __attribute__((ext_vector_type(8)))  __bf16 v8bf;
typedef __attribute__((ext_vector_type(8)))  float  v8f;
typedef int v4i32 __attribute__((vector_size(16)));

#if defined(__has_builtin)
#if __has_builtin(__builtin_amdgcn_global_load_async_to_lds_b128)
#define HAVE_ASYNC_LDS 1
#endif
#endif
#ifndef HAVE_ASYNC_LDS
#define HAVE_ASYNC_LDS 0
#endif

__device__ __forceinline__ __bf16 f2bf(float f) {
    union { float f; uint32_t u; } in; in.f = f;
    uint32_t r = in.u + 0x7FFFu + ((in.u >> 16) & 1u); // round-to-nearest-even
    union { uint16_t u; __bf16 b; } out; out.u = (uint16_t)(r >> 16);
    return out.b;
}

__device__ __forceinline__ v8f vzero8() {
    v8f v;
    #pragma unroll
    for (int i = 0; i < 8; ++i) v[i] = 0.0f;
    return v;
}

#if HAVE_ASYNC_LDS
__device__ __forceinline__ void async_ld16(const __bf16* g, __bf16* l) {
    __builtin_amdgcn_global_load_async_to_lds_b128(
        (__attribute__((address_space(1))) v4i32*)g,
        (__attribute__((address_space(3))) v4i32*)l, 0, 0);
}
__device__ __forceinline__ void wait_async0() {
#if __has_builtin(__builtin_amdgcn_s_wait_asynccnt)
    __builtin_amdgcn_s_wait_asynccnt(0);
#else
    asm volatile("s_wait_asynccnt 0x0" ::: "memory");
#endif
}
#endif

// ---------------------------------------------------------------------------
// Embedding: x[row, c] = wte[idx[row], c] + wpe[t, c]   (fp32 residual stream)
// ---------------------------------------------------------------------------
__global__ __launch_bounds__(256) void embed_kernel(
    const int* __restrict__ idx, const float* __restrict__ wte,
    const float* __restrict__ wpe, float* __restrict__ x) {
    const int row = blockIdx.x;            // b*T + t
    const int t   = row & (TT - 1);
    const int tok = idx[row];
    for (int c = threadIdx.x; c < CC; c += blockDim.x) {
        x[(size_t)row * CC + c] = wte[(size_t)tok * CC + c] + wpe[(size_t)t * CC + c];
    }
}

// ---------------------------------------------------------------------------
// Weight convert+transpose: in fp32 [K][N] -> out bf16 [N][K]
// ---------------------------------------------------------------------------
__global__ __launch_bounds__(256) void transpose_bf16(
    const float* __restrict__ in, __bf16* __restrict__ out, int K, int N) {
    __shared__ float tile[32][33];
    const int tx = threadIdx.x & 31, ty = threadIdx.x >> 5;
    const int k0 = blockIdx.y * 32, n0 = blockIdx.x * 32;
    #pragma unroll
    for (int r = ty; r < 32; r += 8)
        tile[r][tx] = in[(size_t)(k0 + r) * N + n0 + tx];
    __syncthreads();
    #pragma unroll
    for (int r = ty; r < 32; r += 8)
        out[(size_t)(n0 + r) * K + k0 + tx] = f2bf(tile[tx][r]);
}

// Elementwise fp32 -> bf16 (for wte); count divisible by 4.
__global__ __launch_bounds__(256) void convert_bf16(
    const float* __restrict__ in, __bf16* __restrict__ out, long long n4) {
    const long long i = ((long long)blockIdx.x * 256 + threadIdx.x) * 4;
    if (i >= n4 * 4) return;
    float4 v = *(const float4*)(in + i);
    out[i + 0] = f2bf(v.x); out[i + 1] = f2bf(v.y);
    out[i + 2] = f2bf(v.z); out[i + 3] = f2bf(v.w);
}

// ---------------------------------------------------------------------------
// LayerNorm over C=768: fp32 in, bf16 out (feeds WMMA A operands)
// ---------------------------------------------------------------------------
__global__ __launch_bounds__(256) void ln_kernel(
    const float* __restrict__ x, const float* __restrict__ g,
    const float* __restrict__ b, __bf16* __restrict__ out) {
    const int row = blockIdx.x;
    const float* xr = x + (size_t)row * CC;
    float v[3];
    float s = 0.f, ss = 0.f;
    #pragma unroll
    for (int i = 0; i < 3; ++i) {
        v[i] = xr[threadIdx.x + i * 256];
        s += v[i]; ss += v[i] * v[i];
    }
    #pragma unroll
    for (int m = 1; m < 32; m <<= 1) {
        s  += __shfl_xor(s,  m);
        ss += __shfl_xor(ss, m);
    }
    __shared__ float rs[8], rss[8];
    const int lane = threadIdx.x & 31, w = threadIdx.x >> 5;
    if (lane == 0) { rs[w] = s; rss[w] = ss; }
    __syncthreads();
    s = 0.f; ss = 0.f;
    #pragma unroll
    for (int i = 0; i < 8; ++i) { s += rs[i]; ss += rss[i]; }
    const float mu   = s / (float)CC;
    const float var  = ss / (float)CC - mu * mu;
    const float rstd = rsqrtf(var + 1e-3f);
    #pragma unroll
    for (int i = 0; i < 3; ++i) {
        const int c = threadIdx.x + i * 256;
        out[(size_t)row * CC + c] = f2bf((v[i] - mu) * rstd * g[c] + b[c]);
    }
}

// ---------------------------------------------------------------------------
// GEMM: A bf16 [M][K] x Bt bf16 [N][K] (+bias). 256 threads / 8 waves.
// Tile 128(M) x 64(N), K-step 32, double-buffered LDS.
// Preferred path: GLOBAL_LOAD_ASYNC_TO_LDS_B128 (ASYNCcnt) DMAs the next
// K-step tile while WMMAs consume the current buffer; issue-after-barrier
// makes one barrier per step sufficient. Fallback: register prefetch pipeline.
//   mode 0: outF = A@B + bias                      (fp32; logits)
//   mode 1: outF = res + A@B + bias                (fp32; residual adds)
//   mode 2: outB = bf16(gelu(A@B + bias))          (MLP hidden)
//   mode 3: qkv split: cols [0,2C) -> outB[row][2C] (q|k),
//           cols [2C,3C) -> vT[(col-2C)][row]       (V transposed, bf16)
// ---------------------------------------------------------------------------
__global__ __launch_bounds__(256) void gemm_bf16(
    const __bf16* __restrict__ A, const __bf16* __restrict__ Bt,
    const float* __restrict__ bias, const float* __restrict__ res,
    float* __restrict__ outF, __bf16* __restrict__ outB, __bf16* __restrict__ vT,
    int M, int N, int K, int mode) {
    __shared__ __attribute__((aligned(32))) __bf16 ldsA[2][8][32][16];
    __shared__ __attribute__((aligned(32))) __bf16 ldsB[2][4][32][16];

    const int tid  = threadIdx.x;
    const int lane = tid & 31;
    const int w    = tid >> 5;
    const int wmp  = w >> 1;            // M-subtile pair 0..3
    const int wn   = w & 1;             // N-subtile pair 0..1
    const int row0 = blockIdx.y * 128;
    const int col0 = blockIdx.x * 64;

    // Staging: A = 2 chunks/thread (frags f and f+4), B = 1 chunk/thread.
    const int sfrag = tid >> 6;         // 0..3
    const int slane = (tid >> 1) & 31;
    const int shc   = tid & 1;
    const int sG    = slane >> 4;
    const int aKoff = sG * 8 + shc * 16;      // A: K = g*8 + (h&7) + (h>=8)*16
    const int bKoff = sG * 8 + shc * 16;      // B: K = g*8 + (h&7) + (h>=8)*16
    const int aRow0 = row0 + sfrag * 16 + (slane & 15);
    const int aRow1 = aRow0 + 64;             // frag sfrag+4
    int bCol = col0 + sfrag * 16 + (slane & 15);
    if (bCol >= N) bCol = N - 1;              // clamp: epilogue skips these cols

    const __bf16* aSrc0 = A + (size_t)aRow0 * K + aKoff;
    const __bf16* aSrc1 = A + (size_t)aRow1 * K + aKoff;
    const __bf16* bSrc  = Bt + (size_t)bCol * K + bKoff;

    v8f acc00 = vzero8(), acc01 = vzero8(), acc10 = vzero8(), acc11 = vzero8();

    const int m0 = wmp * 2, m1 = m0 + 1;
    const int n0 = wn * 2,  n1 = n0 + 1;

    int buf = 0;
#if HAVE_ASYNC_LDS
    // Prologue: DMA first K-step straight into LDS buffer 0.
    async_ld16(aSrc0, &ldsA[0][sfrag][slane][shc * 8]);
    async_ld16(aSrc1, &ldsA[0][sfrag + 4][slane][shc * 8]);
    async_ld16(bSrc,  &ldsB[0][sfrag][slane][shc * 8]);
    for (int k0 = 0; k0 < K; k0 += 32) {
        wait_async0();        // own fills of lds[buf] complete
        __syncthreads();      // everyone's fills visible; prior reads of nxt done
        const int kn = k0 + 32;
        if (kn < K) {         // DMA next tile while computing (safe post-barrier)
            async_ld16(aSrc0 + kn, &ldsA[buf ^ 1][sfrag][slane][shc * 8]);
            async_ld16(aSrc1 + kn, &ldsA[buf ^ 1][sfrag + 4][slane][shc * 8]);
            async_ld16(bSrc + kn,  &ldsB[buf ^ 1][sfrag][slane][shc * 8]);
        }
        v16bf a0 = *(const v16bf*)&ldsA[buf][m0][lane][0];
        v16bf a1 = *(const v16bf*)&ldsA[buf][m1][lane][0];
        v16bf b0 = *(const v16bf*)&ldsB[buf][n0][lane][0];
        v16bf b1 = *(const v16bf*)&ldsB[buf][n1][lane][0];
        acc00 = __builtin_amdgcn_wmma_f32_16x16x32_bf16(false, a0, false, b0, (short)0, acc00, false, false);
        acc01 = __builtin_amdgcn_wmma_f32_16x16x32_bf16(false, a0, false, b1, (short)0, acc01, false, false);
        acc10 = __builtin_amdgcn_wmma_f32_16x16x32_bf16(false, a1, false, b0, (short)0, acc10, false, false);
        acc11 = __builtin_amdgcn_wmma_f32_16x16x32_bf16(false, a1, false, b1, (short)0, acc11, false, false);
        buf ^= 1;
    }
#else
    // Fallback: register-prefetch software pipeline.
    uint4 ra0 = *(const uint4*)aSrc0;
    uint4 ra1 = *(const uint4*)aSrc1;
    uint4 rb  = *(const uint4*)bSrc;
    for (int k0 = 0; k0 < K; k0 += 32) {
        *(uint4*)&ldsA[buf][sfrag][slane][shc * 8]     = ra0;
        *(uint4*)&ldsA[buf][sfrag + 4][slane][shc * 8] = ra1;
        *(uint4*)&ldsB[buf][sfrag][slane][shc * 8]     = rb;
        __syncthreads();
        const int kn = k0 + 32;
        if (kn < K) {
            ra0 = *(const uint4*)(aSrc0 + kn);
            ra1 = *(const uint4*)(aSrc1 + kn);
            rb  = *(const uint4*)(bSrc + kn);
        }
        v16bf a0 = *(const v16bf*)&ldsA[buf][m0][lane][0];
        v16bf a1 = *(const v16bf*)&ldsA[buf][m1][lane][0];
        v16bf b0 = *(const v16bf*)&ldsB[buf][n0][lane][0];
        v16bf b1 = *(const v16bf*)&ldsB[buf][n1][lane][0];
        acc00 = __builtin_amdgcn_wmma_f32_16x16x32_bf16(false, a0, false, b0, (short)0, acc00, false, false);
        acc01 = __builtin_amdgcn_wmma_f32_16x16x32_bf16(false, a0, false, b1, (short)0, acc01, false, false);
        acc10 = __builtin_amdgcn_wmma_f32_16x16x32_bf16(false, a1, false, b0, (short)0, acc10, false, false);
        acc11 = __builtin_amdgcn_wmma_f32_16x16x32_bf16(false, a1, false, b1, (short)0, acc11, false, false);
        buf ^= 1;
    }
#endif

    // ---- epilogue ----
    #pragma unroll
    for (int m = 0; m < 2; ++m) {
        const int rbase = row0 + (wmp * 2 + m) * 16 + ((lane >> 4) << 3);
        #pragma unroll
        for (int t = 0; t < 2; ++t) {
            v8f acc = m ? (t ? acc11 : acc10) : (t ? acc01 : acc00);
            const int col = col0 + (wn * 2 + t) * 16 + (lane & 15);
            if (col >= N) continue;
            const float bval = bias ? bias[col] : 0.0f;
            if (mode == 3) {
                if (col < 2 * CC) {
                    #pragma unroll
                    for (int vr = 0; vr < 8; ++vr)
                        outB[(size_t)(rbase + vr) * (2 * CC) + col] = f2bf(acc[vr] + bval);
                } else {
                    union { __bf16 h[8]; float4 f4; } tmp;
                    #pragma unroll
                    for (int vr = 0; vr < 8; ++vr) tmp.h[vr] = f2bf(acc[vr] + bval);
                    *(float4*)(vT + (size_t)(col - 2 * CC) * MTOK + rbase) = tmp.f4;
                }
            } else if (mode == 2) {
                #pragma unroll
                for (int vr = 0; vr < 8; ++vr) {
                    float u = acc[vr] + bval;
                    float c = 0.7978845608028654f * (u + 0.044715f * u * u * u);
                    float e = __expf(2.0f * c);              // tanh(c) = (e-1)/(e+1)
                    float v = 0.5f * u * (1.0f + (e - 1.0f) / (e + 1.0f));
                    outB[(size_t)(rbase + vr) * N + col] = f2bf(v);
                }
            } else if (mode == 1) {
                #pragma unroll
                for (int vr = 0; vr < 8; ++vr) {
                    const size_t o = (size_t)(rbase + vr) * N + col;
                    outF[o] = res[o] + acc[vr] + bval;
                }
            } else {
                #pragma unroll
                for (int vr = 0; vr < 8; ++vr)
                    outF[(size_t)(rbase + vr) * N + col] = acc[vr] + bval;
            }
        }
    }
}

// ---------------------------------------------------------------------------
// Fused causal attention, flash-style. qk bf16 [token][2C], vT bf16 [C][MTOK].
// Block = (64 rows, head, batch), 4 fully independent waves (no block
// barriers: the P-matrix LDS bounce is wave-private; LDS ops of one wave
// complete in order, so a wave-level scheduling barrier suffices).
// ---------------------------------------------------------------------------
__global__ __launch_bounds__(128) void attn_fused(
    const __bf16* __restrict__ qk, const __bf16* __restrict__ vT,
    __bf16* __restrict__ y) {
    __shared__ float pS[4][16][32];

    const int tid  = threadIdx.x;
    const int lane = tid & 31;
    const int w    = tid >> 5;
    const int b    = blockIdx.z;
    const int head = blockIdx.y;
    const int i0   = blockIdx.x * 64;
    const int g    = lane >> 4;
    const int r    = lane & 15;
    const float scale = 0.125f; // 1/sqrt(64)

    // ---- Q fragments in registers for all j tiles ----
    v16bf qa[2];
    const int qrow = b * TT + i0 + w * 16 + r;
    #pragma unroll
    for (int ks = 0; ks < 2; ++ks) {
        const __bf16* base = qk + (size_t)qrow * (2 * CC) + head * DD + ks * 32 + g * 16;
        v8bf lo = *(const v8bf*)base;
        v8bf hi = *(const v8bf*)(base + 8);
        qa[ks] = __builtin_shufflevector(lo, hi, 0, 1, 2, 3, 4, 5, 6, 7,
                                         8, 9, 10, 11, 12, 13, 14, 15);
    }

    v8f acc[4];
    float mrun[8], lrun[8];
    #pragma unroll
    for (int n = 0; n < 4; ++n) acc[n] = vzero8();
    #pragma unroll
    for (int vr = 0; vr < 8; ++vr) { mrun[vr] = -3.0e38f; lrun[vr] = 0.0f; }

    for (int j0 = 0; j0 < i0 + 64; j0 += 32) {
        // ---- S = Q K^T (16x32): K-frags contiguous per lane ----
        v8f s0 = vzero8(), s1 = vzero8();
        #pragma unroll
        for (int ks = 0; ks < 2; ++ks) {
            #pragma unroll
            for (int t = 0; t < 2; ++t) {
                const int key = j0 + t * 16 + r;
                v16bf kb = *(const v16bf*)(qk + (size_t)(b * TT + key) * (2 * CC)
                                           + CC + head * DD + ks * 32 + g * 16);
                if (t == 0)
                    s0 = __builtin_amdgcn_wmma_f32_16x16x32_bf16(false, qa[ks], false, kb, (short)0, s0, false, false);
                else
                    s1 = __builtin_amdgcn_wmma_f32_16x16x32_bf16(false, qa[ks], false, kb, (short)0, s1, false, false);
            }
        }

        // ---- scale + causal mask + online softmax ----
        float fac[8];
        #pragma unroll
        for (int vr = 0; vr < 8; ++vr) {
            const int qi = i0 + w * 16 + vr + g * 8;
            const int jA = j0 + r;
            const int jB = j0 + 16 + r;
            float sa = (jA <= qi) ? s0[vr] * scale : -3.0e38f;
            float sb = (jB <= qi) ? s1[vr] * scale : -3.0e38f;
            float mx = fmaxf(sa, sb);
            mx = fmaxf(mx, __shfl_xor(mx, 1));
            mx = fmaxf(mx, __shfl_xor(mx, 2));
            mx = fmaxf(mx, __shfl_xor(mx, 4));
            mx = fmaxf(mx, __shfl_xor(mx, 8));
            const float mnew = fmaxf(mrun[vr], mx);
            const float f  = __expf(mrun[vr] - mnew);
            const float pa = __expf(sa - mnew);
            const float pb = __expf(sb - mnew);
            s0[vr] = pa; s1[vr] = pb;
            float rsum = pa + pb;
            rsum += __shfl_xor(rsum, 1); rsum += __shfl_xor(rsum, 2);
            rsum += __shfl_xor(rsum, 4); rsum += __shfl_xor(rsum, 8);
            lrun[vr] = lrun[vr] * f + rsum;
            mrun[vr] = mnew;
            fac[vr] = f;
        }
        #pragma unroll
        for (int n = 0; n < 4; ++n)
            #pragma unroll
            for (int vr = 0; vr < 8; ++vr) acc[n][vr] *= fac[vr];

        // ---- P: C-layout -> A-frag layout (wave-private LDS bounce) ----
        #pragma unroll
        for (int vr = 0; vr < 8; ++vr) {
            const int i = vr + g * 8;
            pS[w][i][r]      = s0[vr];
            pS[w][i][16 + r] = s1[vr];
        }
        __builtin_amdgcn_wave_barrier();   // order wave-private LDS RAW
        v16bf pa;
        #pragma unroll
        for (int h = 0; h < 16; ++h) {
            const int k = g * 16 + h;
            pa[h] = f2bf(pS[w][r][k]);
        }
        __builtin_amdgcn_wave_barrier();

        // ---- Y += P @ V: V-frags contiguous per lane from vT ----
        #pragma unroll
        for (int n = 0; n < 4; ++n) {
            v16bf vb = *(const v16bf*)(vT + (size_t)(head * DD + n * 16 + r) * MTOK
                                       + b * TT + j0 + g * 16);
            acc[n] = __builtin_amdgcn_wmma_f32_16x16x32_bf16(false, pa, false, vb, (short)0, acc[n], false, false);
        }
    }

    // ---- normalize, write y (bf16, feeds O-projection A operand) ----
    #pragma unroll
    for (int vr = 0; vr < 8; ++vr) {
        const int qi = i0 + w * 16 + vr + g * 8;
        const float inv = 1.0f / lrun[vr];
        #pragma unroll
        for (int n = 0; n < 4; ++n) {
            const int d = head * DD + n * 16 + r;
            y[(size_t)(b * TT + qi) * CC + d] = f2bf(acc[n][vr] * inv);
        }
    }
}

// ---------------------------------------------------------------------------
extern "C" void kernel_launch(void* const* d_in, const int* in_sizes, int n_in,
                              void* d_out, int out_size, void* d_ws, size_t ws_size,
                              hipStream_t stream) {
    (void)in_sizes; (void)n_in; (void)out_size; (void)ws_size;
    const int*   idx   = (const int*)d_in[0];
    const float* wte   = (const float*)d_in[1];
    const float* wpe   = (const float*)d_in[2];
    const float* ln1_g = (const float*)d_in[3];
    const float* ln1_b = (const float*)d_in[4];
    const float* Wqkv  = (const float*)d_in[5];
    const float* bqkv  = (const float*)d_in[6];
    const float* Wo    = (const float*)d_in[7];
    const float* bo    = (const float*)d_in[8];
    const float* ln2_g = (const float*)d_in[9];
    const float* ln2_b = (const float*)d_in[10];
    const float* Wfc   = (const float*)d_in[11];
    const float* bfc   = (const float*)d_in[12];
    const float* Wp    = (const float*)d_in[13];
    const float* bp    = (const float*)d_in[14];
    const float* lnf_g = (const float*)d_in[15];
    const float* lnf_b = (const float*)d_in[16];
    float* out = (float*)d_out;

    // ---- workspace layout ----
    float*  x      = (float*)d_ws;                       // [M][C] fp32 residual
    __bf16* h_bf   = (__bf16*)(x + (size_t)MTOK * CC);   // [M][C]
    __bf16* qk_bf  = h_bf  + (size_t)MTOK * CC;          // [M][2C]
    __bf16* vT_bf  = qk_bf + (size_t)MTOK * 2 * CC;      // [C][M]
    __bf16* y_bf   = vT_bf + (size_t)CC * MTOK;          // [M][C]
    __bf16* ffh_bf = y_bf  + (size_t)MTOK * CC;          // [M][4C]
    __bf16* wqkvT  = ffh_bf + (size_t)MTOK * 4 * CC;     // L x [3C][C]
    __bf16* woT    = wqkvT + (size_t)LLAYERS * 3 * CC * CC; // L x [C][C]
    __bf16* wfcT   = woT   + (size_t)LLAYERS * CC * CC;     // L x [4C][C]
    __bf16* wpT    = wfcT  + (size_t)LLAYERS * 4 * CC * CC; // L x [C][4C]
    __bf16* wteB   = wpT   + (size_t)LLAYERS * CC * 4 * CC; // [V][C]

    // ---- one-time (per launch) weight convert + transpose to bf16 [N][K] ----
    for (int l = 0; l < LLAYERS; ++l) {
        transpose_bf16<<<dim3(3 * CC / 32, CC / 32), 256, 0, stream>>>(
            Wqkv + (size_t)l * CC * 3 * CC, wqkvT + (size_t)l * 3 * CC * CC, CC, 3 * CC);
        transpose_bf16<<<dim3(CC / 32, CC / 32), 256, 0, stream>>>(
            Wo + (size_t)l * CC * CC, woT + (size_t)l * CC * CC, CC, CC);
        transpose_bf16<<<dim3(4 * CC / 32, CC / 32), 256, 0, stream>>>(
            Wfc + (size_t)l * CC * 4 * CC, wfcT + (size_t)l * 4 * CC * CC, CC, 4 * CC);
        transpose_bf16<<<dim3(CC / 32, 4 * CC / 32), 256, 0, stream>>>(
            Wp + (size_t)l * 4 * CC * CC, wpT + (size_t)l * CC * 4 * CC, 4 * CC, CC);
    }
    {
        const long long n4 = (long long)VV * CC / 4;
        const int nblk = (int)((n4 + 255) / 256);
        convert_bf16<<<nblk, 256, 0, stream>>>(wte, wteB, n4);
    }

    embed_kernel<<<MTOK, 256, 0, stream>>>(idx, wte, wpe, x);

    for (int l = 0; l < LLAYERS; ++l) {
        ln_kernel<<<MTOK, 256, 0, stream>>>(x, ln1_g + l * CC, ln1_b + l * CC, h_bf);
        gemm_bf16<<<dim3(3 * CC / 64, MTOK / 128), 256, 0, stream>>>(
            h_bf, wqkvT + (size_t)l * 3 * CC * CC, bqkv + (size_t)l * 3 * CC,
            nullptr, nullptr, qk_bf, vT_bf, MTOK, 3 * CC, CC, /*mode=*/3);
        attn_fused<<<dim3(TT / 64, HH, BBATCH), 128, 0, stream>>>(qk_bf, vT_bf, y_bf);
        gemm_bf16<<<dim3(CC / 64, MTOK / 128), 256, 0, stream>>>(
            y_bf, woT + (size_t)l * CC * CC, bo + (size_t)l * CC,
            x, x, nullptr, nullptr, MTOK, CC, CC, /*mode=*/1);
        ln_kernel<<<MTOK, 256, 0, stream>>>(x, ln2_g + l * CC, ln2_b + l * CC, h_bf);
        gemm_bf16<<<dim3(4 * CC / 64, MTOK / 128), 256, 0, stream>>>(
            h_bf, wfcT + (size_t)l * 4 * CC * CC, bfc + (size_t)l * 4 * CC,
            nullptr, nullptr, ffh_bf, nullptr, MTOK, 4 * CC, CC, /*mode=*/2);
        gemm_bf16<<<dim3(CC / 64, MTOK / 128), 256, 0, stream>>>(
            ffh_bf, wpT + (size_t)l * CC * 4 * CC, bp + (size_t)l * CC,
            x, x, nullptr, nullptr, MTOK, CC, 4 * CC, /*mode=*/1);
    }

    ln_kernel<<<MTOK, 256, 0, stream>>>(x, lnf_g, lnf_b, h_bf);
    // logits = h @ wte^T ; wte is [V][C] row-major == bf16 Bt layout directly
    gemm_bf16<<<dim3((VV + 63) / 64, MTOK / 128), 256, 0, stream>>>(
        h_bf, wteB, nullptr, nullptr, out, nullptr, nullptr, MTOK, VV, CC, /*mode=*/0);
}
